// RelativeMultiHeadedAttention_10917806866773
// MI455X (gfx1250) — matrix operand order repaired and checked by reading, not compiled, not run on hardware
//
#include <hip/hip_runtime.h>
#include <hip/hip_bf16.h>

// ---------------------------------------------------------------------------
// Transformer-XL relative multi-head attention for MI455X (gfx1250, wave32).
// Matmuls: V_WMMA_F32_16X16X32_BF16 (f32 accumulate, bf16 operands).
// GEMM operand staging: CDNA5 Tensor Data Mover (tensor_load_to_lds) into
// double-buffered LDS tiles, synchronized with s_wait_tensorcnt.
// ---------------------------------------------------------------------------

#define B_  2
#define L_  1024
#define D_  1024
#define H_  16
#define DH  64
#define SROW 1032   // 1024 + 8 pad floats per LDS score row
#define TT  32      // transpose tile

typedef __attribute__((ext_vector_type(16))) __bf16 v16bf;
typedef __attribute__((ext_vector_type(8)))  __bf16 v8bf;
typedef __attribute__((ext_vector_type(8)))  float  v8f;
typedef __attribute__((ext_vector_type(4)))  float  v4f;
typedef __attribute__((ext_vector_type(4)))  unsigned int u32x4;
typedef __attribute__((ext_vector_type(8)))  int i32x8;
typedef __attribute__((ext_vector_type(4)))  int i32x4;

#if defined(__has_builtin)
#if __has_builtin(__builtin_amdgcn_tensor_load_to_lds) && \
    __has_builtin(__builtin_amdgcn_s_wait_tensorcnt)
#define USE_TDM 1
#endif
#endif

static __device__ __forceinline__ unsigned short f32_bf16(float f) {
  unsigned u = __builtin_bit_cast(unsigned, f);
  u += 0x7fffu + ((u >> 16) & 1u);          // round-to-nearest-even
  return (unsigned short)(u >> 16);
}
static __device__ __forceinline__ float bf16_f32(unsigned short s) {
  return __builtin_bit_cast(float, ((unsigned)s) << 16);
}
static __device__ __forceinline__ __bf16 us_bf(unsigned short s) {
  return __builtin_bit_cast(__bf16, s);
}
static __device__ __forceinline__ v8bf ld_v8(const unsigned short* p) {
  return *reinterpret_cast<const v8bf*>(p);   // 16B-aligned by construction
}
static __device__ __forceinline__ v16bf cat16(v8bf lo, v8bf hi) {
  return __builtin_shufflevector(lo, hi, 0, 1, 2, 3, 4, 5, 6, 7,
                                 8, 9, 10, 11, 12, 13, 14, 15);
}
static __device__ __forceinline__ v8f wmma_bf16(v16bf a, v16bf b, v8f c) {
  return __builtin_amdgcn_wmma_f32_16x16x32_bf16(false, a, false, b,
                                                 (short)0, c, false, false);
}

// A operand: 16x32 bf16 tile of a row-major matrix (ISA 7.12.2 A layout):
// lane half h covers K groups {h*8..h*8+7} and {16+h*8..16+h*8+7}.
static __device__ __forceinline__ v16bf load_a_frag(const unsigned short* M,
                                                    int ld, int row0, int k0) {
  int ln = threadIdx.x & 31;
  int r = ln & 15, hl = ln >> 4;
  const unsigned short* p = M + (size_t)(row0 + r) * ld + k0 + hl * 8;
  return cat16(ld_v8(p), ld_v8(p + 16));
}

// B operand from the TRANSPOSE of the logical KxN matrix: Mt is N x K
// row-major, column n of B = row n of Mt -> K groups {h*8..h*8+7}, {16+h*8..}.
static __device__ __forceinline__ v16bf load_bt_frag(const unsigned short* Mt,
                                                     int ld, int k0, int col0) {
  int ln = threadIdx.x & 31;
  int c = ln & 15, hl = ln >> 4;
  const unsigned short* p = Mt + (size_t)(col0 + c) * ld + k0 + hl * 8;
  return cat16(ld_v8(p), ld_v8(p + 16));
}

// Same fragment patterns, sourced from an LDS-staged (rows x 32) bf16 tile.
static __device__ __forceinline__ v16bf lds_a_frag(const unsigned short* S,
                                                   int rowbase) {
  int ln = threadIdx.x & 31;
  int r = ln & 15, hl = ln >> 4;
  const unsigned short* p = S + (rowbase + r) * 32 + hl * 8;
  return cat16(ld_v8(p), ld_v8(p + 16));
}
static __device__ __forceinline__ v16bf lds_b_frag(const unsigned short* S,
                                                   int rowbase) {
  int ln = threadIdx.x & 31;
  int c = ln & 15, hl = ln >> 4;
  const unsigned short* p = S + (rowbase + c) * 32 + hl * 8;
  return cat16(ld_v8(p), ld_v8(p + 16));
}

#if defined(USE_TDM)
// TDM 2D tile load: `rows` rows of 32 bf16 (64 B) from a row-major matrix with
// row stride `stride_elems`, packed dense into LDS at byte offset lds_off.
// D# layout per CDNA5 ISA (group0/group1; groups 2-3
// zero for 2D tiles). All fields are workgroup-uniform -> SGPRs.
static __device__ __forceinline__ void tdm_load_2d(unsigned lds_off,
                                                   const void* gptr,
                                                   unsigned rows,
                                                   unsigned stride_elems) {
  unsigned long long ga = (unsigned long long)(size_t)gptr;
  u32x4 g0 = {
      1u,                                    // count=1 (valid user descriptor)
      lds_off,                               // lds_addr (bytes)
      (unsigned)(ga & 0xffffffffu),          // global_addr[31:0]
      (unsigned)((ga >> 32) & 0x01ffffffu) | (2u << 30)  // addr[56:32], type=2
  };
  i32x8 g1 = {
      (int)(1u << 16),                               // data_size=1 (2B), mask=0
      (int)((32u & 0xffffu) << 16),                  // tensor_dim0 = 32
      (int)((rows & 0xffffu) << 16),                 // tensor_dim1 = rows
      (int)(32u << 16),                              // tile_dim0 = 32
      (int)(rows & 0xffffu),                         // tile_dim1 = rows
      (int)stride_elems,                             // tensor_dim0_stride lo
      0, 0
  };
  i32x4 z = {0, 0, 0, 0};
#if __clang_major__ >= 23
  i32x8 z8 = {};
  __builtin_amdgcn_tensor_load_to_lds(g0, g1, z, z, z8, 0);
#else
  __builtin_amdgcn_tensor_load_to_lds(g0, g1, z, z, 0);
#endif
}
#endif

// ---------------------------------------------------------------------------
// f32 -> bf16 bulk conversion (activations, kept row-major as A operands)
__global__ void cvt_bf16_kernel(const float* __restrict__ src,
                                unsigned short* __restrict__ dst, int n) {
  int i = blockIdx.x * blockDim.x + threadIdx.x;
  if (i < n) dst[i] = f32_bf16(src[i]);
}

// f32 (K x N) -> bf16 transposed (N x K), LDS-tiled for coalescing both ways.
__global__ void cvtT_bf16_kernel(const float* __restrict__ src,
                                 unsigned short* __restrict__ dst,
                                 int K, int N) {
  __shared__ unsigned short tile[TT][TT + 2];
  int n0 = blockIdx.x * TT, k0 = blockIdx.y * TT;
  int tx = threadIdx.x, ty = threadIdx.y;   // block (32, 8)
  for (int r = ty; r < TT; r += 8)
    tile[r][tx] = f32_bf16(src[(size_t)(k0 + r) * N + n0 + tx]);
  __syncthreads();
  for (int r = ty; r < TT; r += 8)
    dst[(size_t)(n0 + r) * K + k0 + tx] = tile[tx][r];
}

// bf16 (B x L x D) -> bf16 (B x D x L), LDS-tiled.
__global__ void transpose_bf16_kernel(const unsigned short* __restrict__ src,
                                      unsigned short* __restrict__ dst) {
  __shared__ unsigned short tile[TT][TT + 2];
  int d0 = blockIdx.x * TT, l0 = blockIdx.y * TT, b = blockIdx.z;
  const unsigned short* s = src + (size_t)b * L_ * D_;
  unsigned short* o = dst + (size_t)b * D_ * L_;
  int tx = threadIdx.x, ty = threadIdx.y;   // block (32, 8)
  for (int r = ty; r < TT; r += 8)
    tile[r][tx] = s[(size_t)(l0 + r) * D_ + d0 + tx];
  __syncthreads();
  for (int r = ty; r < TT; r += 8)
    o[(size_t)(d0 + r) * L_ + l0 + tx] = tile[tx][r];
}

// ---------------------------------------------------------------------------
// C = A(MxK) @ B(KxN) with B supplied TRANSPOSED (Bt is N x K row-major).
// Block = 8 waves, tile 64x128; wave computes a 16x64 strip.
// K-chunks of 32 staged in double-buffered LDS by the Tensor Data Mover
// (wave 0 issues two DMA descriptors per chunk; others sync via barriers),
// WMMA operands then come from ds_load_b128. Fallback: direct global b128.
__global__ void gemm_bf16_kernel(const unsigned short* __restrict__ A,
                                 const unsigned short* __restrict__ Bt,
                                 int M, int N, int K,
                                 unsigned short* __restrict__ outBf,
                                 float* __restrict__ outF,
                                 const float* __restrict__ addC,
                                 const float* __restrict__ bias) {
  int w = threadIdx.x >> 5, ln = threadIdx.x & 31;
  int wm = w & 3, wn = w >> 2;
  int rowB = blockIdx.y * 64, colB = blockIdx.x * 128;
  int row0 = rowB + wm * 16;
  int col0 = colB + wn * 64;
  v8f acc[4] = {};

#if defined(USE_TDM)
  __shared__ unsigned short sA[2][64 * 32];    // 4 KB per buffer
  __shared__ unsigned short sB[2][128 * 32];   // 8 KB per buffer
  const int nk = K / 32;
  auto stage = [&](int kc, int buf) {
    if (w == 0) {
      tdm_load_2d((unsigned)(size_t)&sA[buf][0],
                  &A[(size_t)rowB * K + kc * 32], 64, (unsigned)K);
      tdm_load_2d((unsigned)(size_t)&sB[buf][0],
                  &Bt[(size_t)colB * K + kc * 32], 128, (unsigned)K);
    }
  };
  stage(0, 0);
  for (int kc = 0; kc < nk; ++kc) {
    int buf = kc & 1;
    if (kc + 1 < nk) {
      stage(kc + 1, buf ^ 1);
      if (w == 0) __builtin_amdgcn_s_wait_tensorcnt(2);  // chunk kc landed
    } else {
      if (w == 0) __builtin_amdgcn_s_wait_tensorcnt(0);
    }
    __syncthreads();
    v16bf a = lds_a_frag(sA[buf], wm * 16);
#pragma unroll
    for (int nt = 0; nt < 4; ++nt) {
      v16bf b = lds_b_frag(sB[buf], wn * 64 + nt * 16);
      acc[nt] = wmma_bf16(a, b, acc[nt]);
    }
    __syncthreads();   // compute done before this buffer is re-filled
  }
#else
  for (int kk = 0; kk < K; kk += 32) {
    v16bf a = load_a_frag(A, K, row0, kk);
    if (kk + 64 < K)
      __builtin_prefetch(&A[(size_t)(row0 + (ln & 15)) * K + kk + 64], 0, 1);
#pragma unroll
    for (int nt = 0; nt < 4; ++nt) {
      v16bf b = load_bt_frag(Bt, K, kk, col0 + nt * 16);
      acc[nt] = wmma_bf16(a, b, acc[nt]);
    }
  }
#endif

  int c = ln & 15, hl = ln >> 4;
#pragma unroll
  for (int nt = 0; nt < 4; ++nt) {
#pragma unroll
    for (int v = 0; v < 8; ++v) {
      int row = row0 + hl * 8 + v;
      int col = col0 + nt * 16 + c;
      float x = acc[nt][v];
      if (outF) {
        if (addC) x += addC[(size_t)row * N + col];
        if (bias) x += bias[col];
        outF[(size_t)row * N + col] = x;
      } else {
        outBf[(size_t)row * N + col] = f32_bf16(x);
      }
    }
  }
}

// ---------------------------------------------------------------------------
// d_row[h][m] = pb_h . Qr[m, h-slice];  c_row[b][h][j] = cb_h . k[b,j,h-slice]
__global__ void bias_rows_kernel(const unsigned short* __restrict__ kbf,
                                 const unsigned short* __restrict__ qr,
                                 const float* __restrict__ cb,
                                 const float* __restrict__ pb,
                                 float* __restrict__ c_row,
                                 float* __restrict__ d_row) {
  int idx = blockIdx.x * blockDim.x + threadIdx.x;
  int nd = H_ * L_;
  if (idx < nd) {
    int h = idx / L_, m = idx % L_;
    float s = 0.f;
    for (int t = 0; t < DH; ++t)
      s += pb[h * DH + t] * bf16_f32(qr[(size_t)m * D_ + h * DH + t]);
    d_row[idx] = s;
  } else if (idx < nd + B_ * H_ * L_) {
    int j2 = idx - nd;
    int b = j2 / (H_ * L_), r = j2 % (H_ * L_);
    int h = r / L_, j = r % L_;
    float s = 0.f;
    for (int t = 0; t < DH; ++t)
      s += cb[h * DH + t] * bf16_f32(kbf[((size_t)b * L_ + j) * D_ + h * DH + t]);
    c_row[j2] = s;
  }
}

// ---------------------------------------------------------------------------
// Fused attention: one workgroup per (b, h, 16-query-row block).
// LDS holds one 16 x 1024 f32 score buffer (66 KB of 320 KB/WGP).
//  A) S   = q @ Qr^T                    (WMMA, all 64 col-tiles)
//  B) S'  = shift(S) masked j<=i, + d_row + c_row  (register-staged, in place)
//  C) S'' = S' + q @ k^T                (WMMA chained via C operand)
//  D) row softmax                       (wave-shuffle reductions)
//  E) O   = P @ v, split-K across waves + LDS reduction
__global__ void attn_kernel(const unsigned short* __restrict__ qbf,
                            const unsigned short* __restrict__ kbf,
                            const unsigned short* __restrict__ vT,
                            const unsigned short* __restrict__ qrbf,
                            const float* __restrict__ c_row,
                            const float* __restrict__ d_row,
                            unsigned short* __restrict__ obf) {
  extern __shared__ float buf[];          // 16 * SROW floats
  const int i0 = blockIdx.x * 16;
  const int hh = blockIdx.y, bb = blockIdx.z;
  const int tid = threadIdx.x, w = tid >> 5, ln = tid & 31;
  const int cl = ln & 15, hl = ln >> 4;

  const unsigned short* qb  = qbf + (size_t)bb * L_ * D_;
  const unsigned short* kb  = kbf + (size_t)bb * L_ * D_;
  const unsigned short* vtb = vT + (size_t)bb * D_ * L_;

  // q tile fragments (16 rows x dh=64, two 32-wide K chunks), reused in A & C
  v16bf aq0 = load_a_frag(qb, D_, i0, hh * DH);
  v16bf aq1 = load_a_frag(qb, D_, i0, hh * DH + 32);

  // ---- Phase A: X = q @ Qr^T ------------------------------------------------
  for (int n = 0; n < 8; ++n) {
    int col0 = (w * 8 + n) * 16;
    v16bf b0 = load_bt_frag(qrbf, D_, hh * DH,      col0);
    v16bf b1 = load_bt_frag(qrbf, D_, hh * DH + 32, col0);
    v8f acc = {};
    acc = wmma_bf16(aq0, b0, acc);
    acc = wmma_bf16(aq1, b1, acc);
#pragma unroll
    for (int v = 0; v < 8; ++v) buf[(hl * 8 + v) * SROW + col0 + cl] = acc[v];
  }
  __syncthreads();

  // ---- Phase B: TXL shift + positional/content biases (in place) -----------
  // A_b[i,j] = X[i, L-1-i+j] and A_d[i,j] = d_row[L-1-i+j], both only j<=i
  // (reference trils only the b/d terms); A_c = c_row[j] for all j.
  {
    int r = tid >> 4, seg = tid & 15;
    int ig = i0 + r;
    float rv[64];
#pragma unroll
    for (int e = 0; e < 64; ++e) {
      int j = seg * 64 + e;
      float val = c_row[((size_t)bb * H_ + hh) * L_ + j];
      if (j <= ig) {
        int m = (L_ - 1) - ig + j;
        val += buf[r * SROW + m] + d_row[hh * L_ + m];
      }
      rv[e] = val;
    }
    __syncthreads();
#pragma unroll
    for (int e = 0; e < 64; ++e) buf[r * SROW + seg * 64 + e] = rv[e];
  }
  __syncthreads();

  // ---- Phase C: S += q @ k^T (chained through WMMA C operand) ---------------
  for (int n = 0; n < 8; ++n) {
    int col0 = (w * 8 + n) * 16;
    v8f acc;
#pragma unroll
    for (int v = 0; v < 8; ++v) acc[v] = buf[(hl * 8 + v) * SROW + col0 + cl];
    v16bf b0 = load_bt_frag(kb, D_, hh * DH,      col0);
    v16bf b1 = load_bt_frag(kb, D_, hh * DH + 32, col0);
    acc = wmma_bf16(aq0, b0, acc);
    acc = wmma_bf16(aq1, b1, acc);
#pragma unroll
    for (int v = 0; v < 8; ++v) buf[(hl * 8 + v) * SROW + col0 + cl] = acc[v];
  }
  __syncthreads();

  // ---- Phase D: softmax over full row (wave handles 2 rows) -----------------
  for (int rr = w * 2; rr < w * 2 + 2; ++rr) {
    float m = -1e30f;
    for (int j = ln; j < L_; j += 32) m = fmaxf(m, buf[rr * SROW + j]);
#pragma unroll
    for (int o = 16; o > 0; o >>= 1) m = fmaxf(m, __shfl_xor(m, o, 32));
    float s = 0.f;
    for (int j = ln; j < L_; j += 32) s += __expf(buf[rr * SROW + j] - m);
#pragma unroll
    for (int o = 16; o > 0; o >>= 1) s += __shfl_xor(s, o, 32);
    float inv = 1.f / s;
    for (int j = ln; j < L_; j += 32)
      buf[rr * SROW + j] = __expf(buf[rr * SROW + j] - m) * inv;
  }
  __syncthreads();

  // ---- Phase E: O = P @ v, split-K: wave w -> N-tile w&3, K-half w>>2 -------
  {
    int nt = w & 3, half = w >> 2;
    int dd0 = nt * 16;
    v8f acc = {};
    for (int c2 = 0; c2 < 16; ++c2) {
      int j0 = half * 512 + c2 * 32;
      // A operand: P rows from LDS (ds_load_b128, then cvt to bf16)
      const float* pr = &buf[cl * SROW + j0 + hl * 8];
      v4f f0 = *(const v4f*)(pr);
      v4f f1 = *(const v4f*)(pr + 4);
      v4f f2 = *(const v4f*)(pr + 16);
      v4f f3 = *(const v4f*)(pr + 20);
      v16bf a;
#pragma unroll
      for (int e = 0; e < 4; ++e) {
        a[e]      = us_bf(f32_bf16(f0[e]));
        a[e + 4]  = us_bf(f32_bf16(f1[e]));
        a[e + 8]  = us_bf(f32_bf16(f2[e]));
        a[e + 12] = us_bf(f32_bf16(f3[e]));
      }
      // B operand: v^T rows are contiguous in j -> 2x global_load_b128
      v16bf bfr = load_bt_frag(vtb, L_, j0, hh * DH + dd0);
      acc = wmma_bf16(a, bfr, acc);
    }
    __syncthreads();                       // P dead; reuse LDS for reduction
    if (w >= 4) {
#pragma unroll
      for (int v = 0; v < 8; ++v) buf[(w - 4) * 256 + v * 32 + ln] = acc[v];
    }
    __syncthreads();
    if (w < 4) {
#pragma unroll
      for (int v = 0; v < 8; ++v) acc[v] += buf[w * 256 + v * 32 + ln];
#pragma unroll
      for (int v = 0; v < 8; ++v) {
        int row = i0 + hl * 8 + v;
        int col = hh * DH + dd0 + cl;
        obf[((size_t)bb * L_ + row) * D_ + col] = f32_bf16(acc[v]);
      }
    }
  }
}

// ---------------------------------------------------------------------------
// Row-wise LayerNorm: out = (x - mu) * rsqrt(var + 1e-5) * g + b
__global__ void ln_kernel(const float* __restrict__ res,
                          const float* __restrict__ g,
                          const float* __restrict__ b2,
                          float* __restrict__ out) {
  __shared__ float red[256];
  int row = blockIdx.x, tid = threadIdx.x;
  const float* x = res + (size_t)row * D_;
  float s = 0.f, s2 = 0.f;
  for (int j = tid; j < D_; j += 256) { float v = x[j]; s += v; s2 += v * v; }
  red[tid] = s; __syncthreads();
  for (int o = 128; o > 0; o >>= 1) {
    if (tid < o) red[tid] += red[tid + o];
    __syncthreads();
  }
  float mu = red[0] / (float)D_;
  __syncthreads();
  red[tid] = s2; __syncthreads();
  for (int o = 128; o > 0; o >>= 1) {
    if (tid < o) red[tid] += red[tid + o];
    __syncthreads();
  }
  float var = red[0] / (float)D_ - mu * mu;
  float inv = rsqrtf(var + 1e-5f);
  for (int j = tid; j < D_; j += 256)
    out[(size_t)row * D_ + j] = (x[j] - mu) * inv * g[j] + b2[j];
}

// ---------------------------------------------------------------------------
extern "C" void kernel_launch(void* const* d_in, const int* in_sizes, int n_in,
                              void* d_out, int out_size, void* d_ws, size_t ws_size,
                              hipStream_t stream) {
  const float* E    = (const float*)d_in[0];
  const float* Ev   = (const float*)d_in[1];
  const float* R    = (const float*)d_in[2];
  const float* Wq   = (const float*)d_in[3];
  const float* Wke  = (const float*)d_in[4];
  const float* Wkr  = (const float*)d_in[5];
  const float* Wv   = (const float*)d_in[6];
  const float* cb   = (const float*)d_in[7];
  const float* pb   = (const float*)d_in[8];
  const float* Wo_w = (const float*)d_in[9];
  const float* Wo_b = (const float*)d_in[10];
  const float* ln_g = (const float*)d_in[11];
  const float* ln_b = (const float*)d_in[12];
  (void)in_sizes; (void)n_in; (void)out_size; (void)ws_size;

  char* p = (char*)d_ws;
  auto carve = [&](size_t bytes) -> void* {
    void* r = (void*)p;
    p += (bytes + 255) & ~(size_t)255;
    return r;
  };
  const size_t NE = (size_t)B_ * L_ * D_;   // 2M elems
  const size_t NW = (size_t)D_ * D_;        // 1M elems
  unsigned short* Ebf   = (unsigned short*)carve(NE * 2);
  unsigned short* Evbf  = (unsigned short*)carve(NE * 2);
  unsigned short* Rbf   = (unsigned short*)carve(NW * 2);
  unsigned short* WqT   = (unsigned short*)carve(NW * 2);  // transposed weights
  unsigned short* WkeT  = (unsigned short*)carve(NW * 2);
  unsigned short* WkrT  = (unsigned short*)carve(NW * 2);
  unsigned short* WvT   = (unsigned short*)carve(NW * 2);
  unsigned short* WoT   = (unsigned short*)carve(NW * 2);
  unsigned short* qbf   = (unsigned short*)carve(NE * 2);
  unsigned short* kbf   = (unsigned short*)carve(NE * 2);
  unsigned short* vbf   = (unsigned short*)carve(NE * 2);
  unsigned short* vTbf  = (unsigned short*)carve(NE * 2);
  unsigned short* qrbf  = (unsigned short*)carve(NW * 2);
  unsigned short* obf   = (unsigned short*)carve(NE * 2);
  float*          resf  = (float*)carve(NE * 4);
  float*          drow  = (float*)carve((size_t)H_ * L_ * 4);
  float*          crow  = (float*)carve((size_t)B_ * H_ * L_ * 4);

  // 1) activations -> bf16 row-major; weights -> bf16 TRANSPOSED
  cvt_bf16_kernel<<<(int)((NE + 255) / 256), 256, 0, stream>>>(E, Ebf, (int)NE);
  cvt_bf16_kernel<<<(int)((NE + 255) / 256), 256, 0, stream>>>(Ev, Evbf, (int)NE);
  cvt_bf16_kernel<<<(int)((NW + 255) / 256), 256, 0, stream>>>(R, Rbf, (int)NW);
  dim3 gt(D_ / TT, D_ / TT), bt(TT, 8);
  cvtT_bf16_kernel<<<gt, bt, 0, stream>>>(Wq,   WqT,  D_, D_);
  cvtT_bf16_kernel<<<gt, bt, 0, stream>>>(Wke,  WkeT, D_, D_);
  cvtT_bf16_kernel<<<gt, bt, 0, stream>>>(Wkr,  WkrT, D_, D_);
  cvtT_bf16_kernel<<<gt, bt, 0, stream>>>(Wv,   WvT,  D_, D_);
  cvtT_bf16_kernel<<<gt, bt, 0, stream>>>(Wo_w, WoT,  D_, D_);

  // 2) projections (WMMA GEMMs, TDM-staged): q, k, v on B*L rows; Qr on L rows
  dim3 gq(D_ / 128, (B_ * L_) / 64);
  gemm_bf16_kernel<<<gq, 256, 0, stream>>>(Ebf,  WqT,  B_ * L_, D_, D_, qbf, nullptr, nullptr, nullptr);
  gemm_bf16_kernel<<<gq, 256, 0, stream>>>(Evbf, WkeT, B_ * L_, D_, D_, kbf, nullptr, nullptr, nullptr);
  gemm_bf16_kernel<<<gq, 256, 0, stream>>>(Evbf, WvT,  B_ * L_, D_, D_, vbf, nullptr, nullptr, nullptr);
  dim3 gr(D_ / 128, L_ / 64);
  gemm_bf16_kernel<<<gr, 256, 0, stream>>>(Rbf,  WkrT, L_, D_, D_, qrbf, nullptr, nullptr, nullptr);

  // 2b) v -> v^T (per batch) so PV B-operand loads are contiguous
  dim3 gv(D_ / TT, L_ / TT, B_);
  transpose_bf16_kernel<<<gv, bt, 0, stream>>>(vbf, vTbf);

  // 3) global content/position bias rows
  int nb = H_ * L_ + B_ * H_ * L_;
  bias_rows_kernel<<<(nb + 255) / 256, 256, 0, stream>>>(kbf, qrbf, cb, pb, crow, drow);

  // 4) fused attention (WMMA scores + shift + softmax + WMMA PV)
  dim3 ga(L_ / 16, H_, B_);
  attn_kernel<<<ga, 256, 16 * SROW * sizeof(float), stream>>>(
      qbf, kbf, vTbf, qrbf, crow, drow, obf);

  // 5) output projection with fused residual + bias (f32 out)
  gemm_bf16_kernel<<<gq, 256, 0, stream>>>(obf, WoT, B_ * L_, D_, D_,
                                           nullptr, resf, Ev, Wo_b);

  // 6) LayerNorm -> d_out
  ln_kernel<<<B_ * L_, 256, 0, stream>>>(resf, ln_g, ln_b, (float*)d_out);
}
